// SimGNNTensorized_87694642250024
// MI455X (gfx1250) — matrix operand order repaired and checked by reading, not compiled, not run on hardware
//
#include <hip/hip_runtime.h>
#include <math.h>

// ---- Problem constants (match reference) ----
#define NT    65536   // B*N nodes per side
#define NB    256     // graph pairs
#define NN    256     // nodes per graph
#define DD    32      // input feature dim
#define F1D   128
#define F2D   64
#define F3D   64
#define TTD   16      // tensor neurons
#define NBINS 16

typedef _Float16 v16h __attribute__((ext_vector_type(16)));
typedef _Float16 v8h  __attribute__((ext_vector_type(8)));
typedef float    v8f  __attribute__((ext_vector_type(8)));

// ---------------------------------------------------------------------------
// Utility kernels
// ---------------------------------------------------------------------------
__global__ void k_zero(float* __restrict__ p, int n) {
  int i = blockIdx.x * 256 + threadIdx.x;
  if (i < n) p[i] = 0.f;
}

__global__ void k_deg(float* __restrict__ deg, const int* __restrict__ dst, int ne) {
  int e = blockIdx.x * 256 + threadIdx.x;
  if (e < ne) atomicAdd(&deg[dst[e]], 1.0f);
}

// deg includes +1 self-loop -> always >= 1, so dinv = rsqrt(deg+1)
__global__ void k_dinv(float* __restrict__ d, int n) {
  int i = blockIdx.x * 256 + threadIdx.x;
  if (i < n) d[i] = rsqrtf(d[i] + 1.0f);
}

// f32 -> f16 conversion (for async LDS staging of the pair-GEMM inputs)
__global__ void k_cvt_f16(_Float16* __restrict__ o, const float* __restrict__ in, int n) {
  int t = blockIdx.x * 256 + threadIdx.x;
  if (t < n) o[t] = (_Float16)in[t];
}

// ---------------------------------------------------------------------------
// WMMA GEMM: C[M,Nout] = A[M,K] @ W[K,Nout]  (f32 in/out, f16 WMMA compute)
// One wave per 16x16 output tile, 8 waves per block. K multiple of 32.
// Per-lane A fragment = two contiguous 8-float runs -> 4x b128 loads.
// ---------------------------------------------------------------------------
__global__ void k_gemm(const float* __restrict__ A, const float* __restrict__ W,
                       float* __restrict__ C, int M, int K, int Nout) {
  const int lane = threadIdx.x & 31;
  const int wave = threadIdx.x >> 5;
  const int mtiles = M >> 4;
  const int tile = blockIdx.x * 8 + wave;
  const int rt = tile % mtiles;
  const int ct = tile / mtiles;
  const int row = rt * 16 + (lane & 15);
  const int col = ct * 16 + (lane & 15);
  const int off0 = (lane & 16) ? 8 : 0;   // K offset of this half-wave's A chunk

  v8f acc = {};
  for (int k0 = 0; k0 < K; k0 += 32) {
    // A fragment: halfs j=0..7 hold K = off0+0..7, j=8..15 hold K = off0+16..23
    const float* ap = A + (size_t)row * K + k0 + off0;
    float4 x0 = *(const float4*)(ap);
    float4 x1 = *(const float4*)(ap + 4);
    float4 x2 = *(const float4*)(ap + 16);
    float4 x3 = *(const float4*)(ap + 20);
    float af[16];
    ((float4*)af)[0] = x0; ((float4*)af)[1] = x1;
    ((float4*)af)[2] = x2; ((float4*)af)[3] = x3;
    v16h a, b;
#pragma unroll
    for (int j = 0; j < 16; ++j) a[j] = (_Float16)af[j];
    // B fragment: column gather from W (small, L2-resident)
#pragma unroll
    for (int j = 0; j < 16; ++j) {
      int kb = ((lane & 16) ? 8 : 0) + (j < 8 ? j : j + 8);
      b[j] = (_Float16)W[(size_t)(k0 + kb) * Nout + col];
    }
    acc = __builtin_amdgcn_wmma_f32_16x16x32_f16(false, a, false, b,
                                                 (short)0, acc, false, false);
  }
#pragma unroll
  for (int r = 0; r < 8; ++r) {
    int orow = rt * 16 + r + ((lane & 16) ? 8 : 0);
    C[(size_t)orow * Nout + col] = acc[r];
  }
}

// ---------------------------------------------------------------------------
// GCN edge aggregation: out[dst] += h[src] * dinv[src]*dinv[dst]
// grid = (ne/256, F/4); float atomics (global_atomic_add_f32)
// ---------------------------------------------------------------------------
__global__ void k_scatter(float* __restrict__ out, const float* __restrict__ h,
                          const int* __restrict__ src, const int* __restrict__ dst,
                          const float* __restrict__ dinv, int F) {
  int e = blockIdx.x * 256 + threadIdx.x;
  int f = blockIdx.y * 4;
  int s = src[e], d = dst[e];
  float coef = dinv[s] * dinv[d];
  const float* hp = h + (size_t)s * F + f;
  float* op = out + (size_t)d * F + f;
  atomicAdd(op + 0, hp[0] * coef);
  atomicAdd(op + 1, hp[1] * coef);
  atomicAdd(op + 2, hp[2] * coef);
  atomicAdd(op + 3, hp[3] * coef);
}

// self-loop term + bias (+ optional relu):  dst = act(agg + h*dinv^2 + b)
__global__ void k_finish(float* __restrict__ dstp, const float* __restrict__ agg,
                         const float* __restrict__ h, const float* __restrict__ dinv,
                         const float* __restrict__ bias, int F, int relu) {
  int i = blockIdx.x * 256 + threadIdx.x;   // over NT*F (exact multiple)
  int node = i / F, f = i % F;
  float di = dinv[node];
  float v = agg[i] + h[i] * di * di + bias[f];
  if (relu) v = fmaxf(v, 0.f);
  dstp[i] = v;
}

// ---------------------------------------------------------------------------
// Attention pooling, one block (256 thr) per graph
// ---------------------------------------------------------------------------
__global__ void k_attpool(const float* __restrict__ emb, const float* __restrict__ aw,
                          const int* __restrict__ sizes, float* __restrict__ eo) {
  __shared__ float spart[256];
  __shared__ float ssum[F3D];
  __shared__ float sctx[F3D];
  __shared__ float ssig[NN];
  const int b = blockIdx.x, tid = threadIdx.x;
  const int f = tid & 63, part = tid >> 6;
  const float* e = emb + (size_t)b * NN * F3D;

  float acc = 0.f;
  for (int n = part * 64; n < part * 64 + 64; ++n) acc += e[n * F3D + f];
  spart[tid] = acc;
  __syncthreads();
  if (tid < 64)
    ssum[tid] = spart[tid] + spart[64 + tid] + spart[128 + tid] + spart[192 + tid];
  __syncthreads();
  if (tid < 64) {
    float a = 0.f;
    for (int i = 0; i < F3D; ++i) a += ssum[i] * aw[i * F3D + tid];
    sctx[tid] = tanhf(a / (float)sizes[b]);
  }
  __syncthreads();
  {
    int n = tid;
    float d = 0.f;
    for (int i = 0; i < F3D; ++i) d += e[n * F3D + i] * sctx[i];
    ssig[n] = 1.f / (1.f + expf(-d));
  }
  __syncthreads();
  float acc2 = 0.f;
  for (int n = part * 64; n < part * 64 + 64; ++n) acc2 += e[n * F3D + f] * ssig[n];
  spart[tid] = acc2;
  __syncthreads();
  if (tid < 64)
    eo[b * F3D + tid] = spart[tid] + spart[64 + tid] + spart[128 + tid] + spart[192 + tid];
}

// ---------------------------------------------------------------------------
// NTN: score[b,t] = relu(e1.A_t.e2 + [e1,e2].Bw[:,t] + bias[t])
// ---------------------------------------------------------------------------
__global__ void k_ntn(const float* __restrict__ e1, const float* __restrict__ e2,
                      const float* __restrict__ A, const float* __restrict__ Bw,
                      const float* __restrict__ bias, float* __restrict__ outn) {
  __shared__ float s1[F3D], s2[F3D], sp[256];
  const int b = blockIdx.x, tid = threadIdx.x;
  if (tid < 64) { s1[tid] = e1[b * 64 + tid]; s2[tid] = e2[b * 64 + tid]; }
  __syncthreads();
  const int t = tid & 15, part = tid >> 4;
  float acc = 0.f;
  for (int ii = 0; ii < 4; ++ii) {
    int i = part * 4 + ii;
    float w = 0.f;
    for (int j = 0; j < 64; ++j) w += A[((size_t)t * 64 + i) * 64 + j] * s2[j];
    acc += s1[i] * w;
  }
  sp[tid] = acc;
  __syncthreads();
  if (tid < 16) {
    float s = 0.f;
    for (int p = 0; p < 16; ++p) s += sp[p * 16 + tid];
    for (int i = 0; i < 64; ++i)
      s += s1[i] * Bw[i * 16 + tid] + s2[i] * Bw[(64 + i) * 16 + tid];
    s += bias[tid];
    outn[b * 16 + tid] = fmaxf(s, 0.f);
  }
}

// ---------------------------------------------------------------------------
// Pair histogram via WMMA. One block per pair; q_b, c_b (pre-converted f16)
// staged into LDS with async Global->LDS b128 copies (ASYNCcnt path).
// Pass 1: per-pair min/max of S = q@c^T. Pass 2: recompute + bin.
// ---------------------------------------------------------------------------
__device__ inline v8f hist_tile(const _Float16* qh, const _Float16* ch,
                                int ti, int tj, int lane) {
  v8f acc = {};
  const int ar = (ti * 16 + (lane & 15)) * F3D + ((lane & 16) ? 8 : 0);
  const int br = (tj * 16 + (lane & 15)) * F3D + ((lane & 16) ? 8 : 0);
#pragma unroll
  for (int ks = 0; ks < 2; ++ks) {
    const int k0 = ks * 32;
    const v8h* pa = (const v8h*)(qh + ar + k0);
    v8h alo = pa[0], ahi = pa[2];          // two 16B runs, 32 halfs apart/2
    const v8h* pb = (const v8h*)(ch + br + k0);
    v8h blo = pb[0], bhi = pb[2];          // two 16B runs, 32 halfs apart/2
    v16h a  = __builtin_shufflevector(alo, ahi, 0,1,2,3,4,5,6,7,8,9,10,11,12,13,14,15);
    v16h bm = __builtin_shufflevector(blo, bhi, 0,1,2,3,4,5,6,7,8,9,10,11,12,13,14,15);
    acc = __builtin_amdgcn_wmma_f32_16x16x32_f16(false, a, false, bm,
                                                 (short)0, acc, false, false);
  }
  return acc;
}

__global__ void k_hist(const _Float16* __restrict__ qg, const _Float16* __restrict__ cg,
                       float* __restrict__ hist_out) {
  extern __shared__ char smem[];
  _Float16* qh = (_Float16*)smem;            // 256*64 f16 = 32 KB
  _Float16* ch = qh + NN * F3D;              // 32 KB
  float* red = (float*)(ch + NN * F3D);      // [0..7]=min/wave [8..15]=max/wave [16],[17]=global
  float* hh  = red + 24;                     // 16 bins
  const int b = blockIdx.x, tid = threadIdx.x;
  const int lane = tid & 31, wave = tid >> 5;

  // Async Global->LDS staging: 16 B per lane per op, 2048 chunks per matrix.
  {
    const unsigned long long gq = (unsigned long long)(uintptr_t)(qg + (size_t)b * NN * F3D);
    const unsigned long long gc = (unsigned long long)(uintptr_t)(cg + (size_t)b * NN * F3D);
    const unsigned int lq = (unsigned int)(uintptr_t)qh;
    const unsigned int lc = (unsigned int)(uintptr_t)ch;
#pragma unroll
    for (int it = 0; it < 8; ++it) {
      const int i = it * 256 + tid;
      unsigned int       la = lq + (unsigned int)i * 16u;
      unsigned long long ga = gq + (unsigned long long)i * 16ull;
      asm volatile("global_load_async_to_lds_b128 %0, %1, off"
                   :: "v"(la), "v"(ga) : "memory");
      unsigned int       lb = lc + (unsigned int)i * 16u;
      unsigned long long gb = gc + (unsigned long long)i * 16ull;
      asm volatile("global_load_async_to_lds_b128 %0, %1, off"
                   :: "v"(lb), "v"(gb) : "memory");
    }
    asm volatile("s_wait_asynccnt 0x0" ::: "memory");
  }
  if (tid < NBINS) hh[tid] = 0.f;
  __syncthreads();

  // Pass 1: min/max
  float vmin = 3.4e38f, vmax = -3.4e38f;
  for (int tile = wave; tile < 256; tile += 8) {
    v8f acc = hist_tile(qh, ch, tile >> 4, tile & 15, lane);
#pragma unroll
    for (int r = 0; r < 8; ++r) { vmin = fminf(vmin, acc[r]); vmax = fmaxf(vmax, acc[r]); }
  }
  for (int off = 16; off >= 1; off >>= 1) {
    vmin = fminf(vmin, __shfl_down(vmin, off, 32));
    vmax = fmaxf(vmax, __shfl_down(vmax, off, 32));
  }
  if (lane == 0) { red[wave] = vmin; red[8 + wave] = vmax; }
  __syncthreads();
  if (tid == 0) {
    float mn = red[0], mx = red[8];
    for (int w = 1; w < 8; ++w) { mn = fminf(mn, red[w]); mx = fmaxf(mx, red[8 + w]); }
    red[16] = mn; red[17] = mx;
  }
  __syncthreads();
  const float mn = red[16];
  const float scale = (float)NBINS / fmaxf(red[17] - mn, 1e-12f);

  // Pass 2: recompute + histogram into LDS (ds_add_f32)
  for (int tile = wave; tile < 256; tile += 8) {
    v8f acc = hist_tile(qh, ch, tile >> 4, tile & 15, lane);
#pragma unroll
    for (int r = 0; r < 8; ++r) {
      int bi = (int)floorf((acc[r] - mn) * scale);
      bi = bi < 0 ? 0 : (bi > NBINS - 1 ? NBINS - 1 : bi);
      atomicAdd(&hh[bi], 1.0f);
    }
  }
  __syncthreads();
  if (tid < NBINS)
    hist_out[b * NBINS + tid] = hh[tid] * (1.0f / ((float)NN * (float)NN));
}

// ---------------------------------------------------------------------------
// Final MLP + GED: one thread per pair
// ---------------------------------------------------------------------------
__global__ void k_final(const float* __restrict__ ntn, const float* __restrict__ hist,
                        const float* __restrict__ w1, const float* __restrict__ b1,
                        const float* __restrict__ w2, const float* __restrict__ b2,
                        const int* __restrict__ qs, const int* __restrict__ cs,
                        float* __restrict__ out) {
  int b = blockIdx.x * 256 + threadIdx.x;
  if (b >= NB) return;
  float in[32];
  for (int i = 0; i < 16; ++i) { in[i] = ntn[b * 16 + i]; in[16 + i] = hist[b * 16 + i]; }
  float a2 = b2[0];
  for (int j = 0; j < 16; ++j) {
    float h = b1[j];
    for (int i = 0; i < 32; ++i) h += in[i] * w1[i * 16 + j];
    h = fmaxf(h, 0.f);
    a2 += h * w2[j];
  }
  float p = 1.f / (1.f + expf(-a2));
  out[b] = -0.5f * (float)(qs[b] + cs[b]) * logf(p);
}

// ---------------------------------------------------------------------------
// Host orchestration
// ---------------------------------------------------------------------------
static void run_side(const float* x, const int* edges, int ne,
                     const float* W1, const float* b1, const float* W2, const float* b2,
                     const float* W3, const float* b3,
                     float* tmpH, float* aggO, float* dinv, float* emb,
                     hipStream_t s) {
  const int* srcp = edges;
  const int* dstp = edges + ne;
  // degree + normalization
  k_zero<<<NT / 256, 256, 0, s>>>(dinv, NT);
  k_deg<<<ne / 256, 256, 0, s>>>(dinv, dstp, ne);
  k_dinv<<<NT / 256, 256, 0, s>>>(dinv, NT);
  // layer 1: D=32 -> 128, relu
  k_gemm<<<(NT / 16) * (F1D / 16) / 8, 256, 0, s>>>(x, W1, tmpH, NT, DD, F1D);
  k_zero<<<NT * F1D / 256, 256, 0, s>>>(aggO, NT * F1D);
  { dim3 g(ne / 256, F1D / 4); k_scatter<<<g, 256, 0, s>>>(aggO, tmpH, srcp, dstp, dinv, F1D); }
  k_finish<<<NT * F1D / 256, 256, 0, s>>>(aggO, aggO, tmpH, dinv, b1, F1D, 1);
  // layer 2: 128 -> 64, relu
  k_gemm<<<(NT / 16) * (F2D / 16) / 8, 256, 0, s>>>(aggO, W2, tmpH, NT, F1D, F2D);
  k_zero<<<NT * F2D / 256, 256, 0, s>>>(aggO, NT * F2D);
  { dim3 g(ne / 256, F2D / 4); k_scatter<<<g, 256, 0, s>>>(aggO, tmpH, srcp, dstp, dinv, F2D); }
  k_finish<<<NT * F2D / 256, 256, 0, s>>>(aggO, aggO, tmpH, dinv, b2, F2D, 1);
  // layer 3: 64 -> 64, no relu, write to emb
  k_gemm<<<(NT / 16) * (F3D / 16) / 8, 256, 0, s>>>(aggO, W3, tmpH, NT, F2D, F3D);
  k_zero<<<NT * F3D / 256, 256, 0, s>>>(aggO, NT * F3D);
  { dim3 g(ne / 256, F3D / 4); k_scatter<<<g, 256, 0, s>>>(aggO, tmpH, srcp, dstp, dinv, F3D); }
  k_finish<<<NT * F3D / 256, 256, 0, s>>>(emb, aggO, tmpH, dinv, b3, F3D, 0);
}

extern "C" void kernel_launch(void* const* d_in, const int* in_sizes, int n_in,
                              void* d_out, int out_size, void* d_ws, size_t ws_size,
                              hipStream_t stream) {
  const float* x_q  = (const float*)d_in[0];
  const int*   e_q  = (const int*)d_in[1];
  const float* x_c  = (const float*)d_in[2];
  const int*   e_c  = (const int*)d_in[3];
  const int*   qs   = (const int*)d_in[4];
  const int*   cs   = (const int*)d_in[5];
  const float* W1   = (const float*)d_in[6];
  const float* b1   = (const float*)d_in[7];
  const float* W2   = (const float*)d_in[8];
  const float* b2   = (const float*)d_in[9];
  const float* W3   = (const float*)d_in[10];
  const float* b3   = (const float*)d_in[11];
  const float* attw = (const float*)d_in[12];
  const float* ntnA = (const float*)d_in[13];
  const float* ntnB = (const float*)d_in[14];
  const float* ntnb = (const float*)d_in[15];
  const float* f1w  = (const float*)d_in[16];
  const float* f1b  = (const float*)d_in[17];
  const float* f2w  = (const float*)d_in[18];
  const float* f2b  = (const float*)d_in[19];
  const int ne = in_sizes[1] / 2;

  // workspace carve-out (~117 MB)
  char* w = (char*)d_ws;
  float*     tmpH  = (float*)w;     w += (size_t)NT * F1D * 4;
  float*     aggO  = (float*)w;     w += (size_t)NT * F1D * 4;
  float*     embq  = (float*)w;     w += (size_t)NT * F3D * 4;
  float*     embc  = (float*)w;     w += (size_t)NT * F3D * 4;
  _Float16*  embqh = (_Float16*)w;  w += (size_t)NT * F3D * 2;
  _Float16*  embch = (_Float16*)w;  w += (size_t)NT * F3D * 2;
  float*     dinvq = (float*)w;     w += (size_t)NT * 4;
  float*     dinvc = (float*)w;     w += (size_t)NT * 4;
  float*     e1    = (float*)w;     w += (size_t)NB * F3D * 4;
  float*     e2    = (float*)w;     w += (size_t)NB * F3D * 4;
  float*     ntnS  = (float*)w;     w += (size_t)NB * TTD * 4;
  float*     histS = (float*)w;     w += (size_t)NB * NBINS * 4;

  run_side(x_q, e_q, ne, W1, b1, W2, b2, W3, b3, tmpH, aggO, dinvq, embq, stream);
  run_side(x_c, e_c, ne, W1, b1, W2, b2, W3, b3, tmpH, aggO, dinvc, embc, stream);

  k_attpool<<<NB, 256, 0, stream>>>(embq, attw, qs, e1);
  k_attpool<<<NB, 256, 0, stream>>>(embc, attw, cs, e2);
  k_ntn<<<NB, 256, 0, stream>>>(e1, e2, ntnA, ntnB, ntnb, ntnS);

  // f16 copies of embeddings for async-LDS staging in the pair-hist kernel
  k_cvt_f16<<<NT * F3D / 256, 256, 0, stream>>>(embqh, embq, NT * F3D);
  k_cvt_f16<<<NT * F3D / 256, 256, 0, stream>>>(embch, embc, NT * F3D);

  const size_t sm = (size_t)NN * F3D * 2 * sizeof(_Float16) + (24 + NBINS) * sizeof(float);
  k_hist<<<NB, 256, sm, stream>>>(embqh, embch, histS);

  k_final<<<1, 256, 0, stream>>>(ntnS, histS, f1w, f1b, f2w, f2b, qs, cs, (float*)d_out);
}
